// Logic_Learning_Model_50440095924921
// MI455X (gfx1250) — matrix operand order repaired and checked by reading, not compiled, not run on hardware
//
#include <hip/hip_runtime.h>
#include <hip/hip_bf16.h>

typedef __attribute__((ext_vector_type(16))) _Float16 v16h;
typedef __attribute__((ext_vector_type(8)))  float    v8f;
typedef __attribute__((ext_vector_type(4)))  unsigned v4u;

#define NE   256        // events per predicate, also Q grid points
#define NP   5
#define TOLF 0.1f
#define RESF 0.3f

union V16U { v16h h; v4u q[2]; };

// One workgroup (256 threads = 8 wave32) per sample.
__global__ __launch_bounds__(256)
void logic_ll_wmma(const float* __restrict__ times,   // [B,5,256] f32 (sorted per pred)
                   const int*   __restrict__ states,  // [B,5,256] i32
                   const float* __restrict__ base,    // [1]
                   const float* __restrict__ weights, // [3]
                   float* __restrict__ out)           // [1]
{
    __shared__ float t0s[NE], t1s[NE], t2s[NE], t3s[NE], ths[NE];
    __shared__ int   s0s[NE], s1s[NE], s2s[NE], s3s[NE], shs[NE];
    __shared__ float e0s[NE], e1s[NE], e2s[NE], e3s[NE];
    __shared__ float tqs[NE], etqs[NE], feat0s[NE], red[NE];
    // R-matrix pre-expanded to f16 B-fragments, fragment-order.
    // frag = nt*8+kt (128 frags). Per lane 16 halves split into two 16B planes
    // (lane stride 16B -> wave load = 512B contiguous, near conflict-free).
    __shared__ v4u BfrLo[128 * 32];   // halves 0..7  of each lane   (64 KB)
    __shared__ v4u BfrHi[128 * 32];   // halves 8..15 of each lane   (64 KB)

    const int tid  = threadIdx.x;
    const int b    = blockIdx.x;
    const int lane = tid & 31;
    const int wave = tid >> 5;
    const int half = lane >> 4;     // 0: lanes 0-15, 1: lanes 16-31
    const int lm   = lane & 15;

    const float* tb = times  + (size_t)b * NP * NE;
    const int*   sb = states + (size_t)b * NP * NE;

    // ---- Load per-sample data + exp tables (exp(76.8) ~ 2.3e33 fits f32) ----
    float tv;
    tv = tb[0*NE + tid]; t0s[tid] = tv; e0s[tid] = __expf(tv);
    tv = tb[1*NE + tid]; t1s[tid] = tv; e1s[tid] = __expf(tv);
    tv = tb[2*NE + tid]; t2s[tid] = tv; e2s[tid] = __expf(tv);
    tv = tb[3*NE + tid]; t3s[tid] = tv; e3s[tid] = __expf(tv);
    ths[tid] = tb[4*NE + tid];
    s0s[tid] = sb[0*NE + tid];
    s1s[tid] = sb[1*NE + tid];
    s2s[tid] = sb[2*NE + tid];
    s3s[tid] = sb[3*NE + tid];
    shs[tid] = sb[4*NE + tid];
    __syncthreads();

    // ---- One-time expansion of R[i,j] = (s0_i&&s1_j&&t0_i<t1_j-tol) into
    //      f16 B-fragments (B-layout 32x16 f16, contiguous k order used here: element e <-> k=e+16*half) ----
    // 4096 fragment-lane entries, 16 per thread, lane-contiguous stores.
    for (int it = 0; it < 16; ++it) {
        const int entry  = it * 256 + tid;
        const int frag   = entry >> 5;          // nt*8 + kt
        const int lane_e = entry & 31;
        const int kt     = frag & 7;
        const int nt     = frag >> 3;
        const int j      = nt * 16 + (lane_e & 15);
        const int khalf  = (lane_e >> 4) * 16;
        const float t1j  = t1s[j] - TOLF;
        const bool  s1j  = (s1s[j] == 1);
        unsigned wlo[4], whi[4];
        #pragma unroll
        for (int p = 0; p < 8; ++p) {
            const int k = 2 * p + khalf + 32 * kt;
            unsigned w = 0u;
            if (s1j && (s0s[k]     == 1) && (t0s[k]     < t1j)) w |= 0x00003C00u;
            if (s1j && (s0s[k + 1] == 1) && (t0s[k + 1] < t1j)) w |= 0x3C000000u;
            if (p < 4) wlo[p] = w; else whi[p - 4] = w;
        }
        BfrLo[entry] = (v4u){wlo[0], wlo[1], wlo[2], wlo[3]};
        BfrHi[entry] = (v4u){whi[0], whi[1], whi[2], whi[3]};
    }

    const float w0 = weights[0], w1 = weights[1], w2 = weights[2], bsv = base[0];
    const int qt0 = wave, qt1 = wave + 8;       // this wave's two q-tiles
    float thread_contrib = 0.0f;

    #pragma unroll 1
    for (int qs = 0; qs < 2; ++qs) {
        // ---- Query times: set0 = head events [1:] (pad q=255), set1 = grid ----
        const float tq = (qs == 0) ? ths[(tid + 1 < NE) ? tid + 1 : NE - 1]
                                   : RESF * (float)tid;
        tqs[tid]  = tq;
        etqs[tid] = __expf(-tq);
        __syncthreads();   // also publishes Bfr* on first iteration

        // ================= WMMA: feat0[q] = a(q)^T R b(q) =================
        // Build all A fragments for BOTH q-tiles (a[q,i] = [t0_i<=tq] e^{t0_i-tq}).
        const float tq_m0  = tqs[qt0 * 16 + lm],  etq_m0 = etqs[qt0 * 16 + lm];
        const float tq_m1  = tqs[qt1 * 16 + lm],  etq_m1 = etqs[qt1 * 16 + lm];
        v16h Af0[8], Af1[8];
        #pragma unroll
        for (int kt = 0; kt < 8; ++kt) {
            #pragma unroll
            for (int e = 0; e < 16; ++e) {
                // A-layout 16x32 f16 (matches B packing): k = e + 16*half
                const int k = e + 16 * half + 32 * kt;
                const float t0v = t0s[k], e0v = e0s[k];
                Af0[kt][e] = (_Float16)((t0v <= tq_m0) ? e0v * etq_m0 : 0.0f);
                Af1[kt][e] = (_Float16)((t0v <= tq_m1) ? e0v * etq_m1 : 0.0f);
            }
        }
        // C-tile rows for this lane: m = qbase + r + 8*half
        float tq_r0[8], etq_r0[8], tq_r1[8], etq_r1[8];
        #pragma unroll
        for (int r = 0; r < 8; ++r) {
            const int m0 = qt0 * 16 + r + 8 * half;
            const int m1 = qt1 * 16 + r + 8 * half;
            tq_r0[r] = tqs[m0]; etq_r0[r] = etqs[m0];
            tq_r1[r] = tqs[m1]; etq_r1[r] = etqs[m1];
        }

        // 2-deep software pipeline over the 128 B fragments: prefetch frag f+1
        // while WMMAs consume frag f (DS completes in order -> small dscnt wait,
        // not a full drain). Parity of f = parity of kt since nt*8 is even, so
        // the buffer state carries cleanly across nt iterations.
        // NOTE: unroll 1 keeps live VGPRs < 256 (no s_set_vgpr_msb thrash) and
        // the loop body small for I$.
        float facc0[8] = {0,0,0,0,0,0,0,0};
        float facc1[8] = {0,0,0,0,0,0,0,0};
        V16U ubuf[2];
        ubuf[0].q[0] = BfrLo[lane];            // frag 0
        ubuf[0].q[1] = BfrHi[lane];
        #pragma unroll 1
        for (int nt = 0; nt < 16; ++nt) {
            const int   j   = nt * 16 + lm;     // B/C column for this lane
            const float t1v = t1s[j];
            const float e1v = e1s[j];
            v8f c0 = {}, c1 = {};
            #pragma unroll
            for (int kt = 0; kt < 8; ++kt) {
                const int f   = nt * 8 + kt;
                const int fn  = (f + 1 < 128) ? f + 1 : 127;   // clamped prefetch
                const int idn = fn * 32 + lane;
                ubuf[(kt + 1) & 1].q[0] = BfrLo[idn];
                ubuf[(kt + 1) & 1].q[1] = BfrHi[idn];
                const v16h Bf = ubuf[kt & 1].h;
                c0 = __builtin_amdgcn_wmma_f32_16x16x32_f16(
                        false, Af0[kt], false, Bf, (short)0, c0, false, false);
                c1 = __builtin_amdgcn_wmma_f32_16x16x32_f16(
                        false, Af1[kt], false, Bf, (short)0, c1, false, false);
            }
            // Fused row-dot with b(q): b[q,j] = [t1_j < tq - tol] e^{t1_j - tq}
            // (overlaps the prefetched first load of the next nt)
            #pragma unroll
            for (int r = 0; r < 8; ++r) {
                const float b0 = (t1v < tq_r0[r] - TOLF) ? e1v * etq_r0[r] : 0.0f;
                const float b1 = (t1v < tq_r1[r] - TOLF) ? e1v * etq_r1[r] : 0.0f;
                facc0[r] += c0[r] * b0;
                facc1[r] += c1[r] * b1;
            }
        }
        // Reduce over the 16 lanes sharing each m (xor 1,2,4,8 stays in half)
        #pragma unroll
        for (int r = 0; r < 8; ++r) {
            float v0 = facc0[r], v1 = facc1[r];
            v0 += __shfl_xor(v0, 1); v0 += __shfl_xor(v0, 2);
            v0 += __shfl_xor(v0, 4); v0 += __shfl_xor(v0, 8);
            v1 += __shfl_xor(v1, 1); v1 += __shfl_xor(v1, 2);
            v1 += __shfl_xor(v1, 4); v1 += __shfl_xor(v1, 8);
            if (lm == 0) {
                feat0s[qt0 * 16 + r + 8 * half] = v0;
                feat0s[qt1 * 16 + r + 8 * half] = v1;
            }
        }

        // ================= Scalar features (O(E) per query) =================
        const float etq = etqs[tid];
        float f1 = 0.0f, f2 = 0.0f;
        int cnt = 0;
        #pragma unroll 1
        for (int e = 0; e < NE; ++e) {
            if (s2s[e] == 1 && t2s[e] < tq - TOLF) f1 += e2s[e] * etq;
            const float t3v = t3s[e];
            if (s3s[e] == 0 && t3v <= tq && (tq - t3v) <= TOLF) f2 += e3s[e] * etq;
            cnt += (tq > ths[e]) ? 1 : 0;
        }
        const int idx     = (cnt - 1) & (NE - 1);      // numpy negative-index wrap
        const int counter = 1 - shs[idx];
        const float eff01 = (counter == 1) ? 1.0f : -1.0f;  // HEAD_SIGNS[0..1]=1
        const float eff2  = (counter == 0) ? 1.0f : -1.0f;  // HEAD_SIGNS[2]=0

        __syncthreads();   // feat0s ready
        const float logit = bsv + w0 * feat0s[tid] * eff01
                                + w1 * f1          * eff01
                                + w2 * f2          * eff2;
        if (qs == 0) thread_contrib += (tid < NE - 1) ? logit : 0.0f;  // log-sum (255 pts)
        else         thread_contrib -= RESF * __expf(logit);           // -integral
        __syncthreads();   // before tqs/feat0s are overwritten
    }

    // ---- Block reduce + global accumulate ----
    red[tid] = thread_contrib;
    __syncthreads();
    for (int s = 128; s > 0; s >>= 1) {
        if (tid < s) red[tid] += red[tid + s];
        __syncthreads();
    }
    if (tid == 0) atomicAdd(out, red[0]);
}

extern "C" void kernel_launch(void* const* d_in, const int* in_sizes, int n_in,
                              void* d_out, int out_size, void* d_ws, size_t ws_size,
                              hipStream_t stream) {
    const float* times   = (const float*)d_in[0];   // [256,5,256]
    const int*   states  = (const int*)  d_in[1];   // [256,5,256]
    const float* base    = (const float*)d_in[2];   // [1]
    const float* weights = (const float*)d_in[3];   // [3]
    float* out = (float*)d_out;

    hipMemsetAsync(out, 0, sizeof(float), stream);  // graph-capture-safe memset node
    logic_ll_wmma<<<256, 256, 0, stream>>>(times, states, base, weights, out);
}
